// SequenceGeneratorStd_7756710937131
// MI455X (gfx1250) — matrix operand order repaired and checked by reading, not compiled, not run on hardware
//
#include <hip/hip_runtime.h>
#include <math.h>

// ---------------- problem constants ----------------
#define VSZ   50257
#define VPAD  50304          // 786 * 64
#define D_    512
#define B_    32
#define K_    5
#define BK_   160            // B_*K_
#define L_    32
#define GC_   34             // gen columns  (L+2)
#define SC_   33             // score columns (L+1)
#define PAD_T 0
#define SOS_T 1
#define EOS_T 2
#define NEGINF (-INFINITY)

typedef __attribute__((ext_vector_type(16))) __bf16 v16bf;
typedef __attribute__((ext_vector_type(8)))  __bf16 v8bf;
typedef __attribute__((ext_vector_type(8)))  float  v8f;

// ---- CDNA5 async global->LDS path (guarded; falls back to load+ds_store) ----
#if defined(__has_builtin)
#if __has_builtin(__builtin_amdgcn_global_load_async_to_lds_b128)
#define HAVE_ASYNC_LDS 1
#endif
#endif

#ifdef HAVE_ASYNC_LDS
// builtin takes generic pointers to 16-byte int vectors (per hipcc diagnostic)
typedef int async_v4i __attribute__((vector_size(16)));
#define ASYNC_PTR(p) ((async_v4i*)(uintptr_t)(p))
#if __has_builtin(__builtin_amdgcn_s_wait_asynccnt)
#define WAIT_ASYNC() __builtin_amdgcn_s_wait_asynccnt(0)
#else
#define WAIT_ASYNC() asm volatile("s_wait_asynccnt 0" ::: "memory")
#endif
#endif

// ---------------- one-time prep kernels ----------------

// w_out is [D][V] row-major f32 -> wt [VPAD][D] row-major bf16 (pad rows = 0)
__global__ __launch_bounds__(256) void convert_wt(const float* __restrict__ w,
                                                  __bf16* __restrict__ wt) {
  size_t idx = (size_t)blockIdx.x * 256 + threadIdx.x;   // idx over d*VPAD + n
  if (idx >= (size_t)D_ * VPAD) return;
  int d = (int)(idx / VPAD);
  int n = (int)(idx % VPAD);
  float v = (n < VSZ) ? w[(size_t)d * VSZ + n] : 0.0f;   // coalesced read over n
  wt[(size_t)n * D_ + d] = (__bf16)v;
}

__global__ __launch_bounds__(256) void ctx_kernel(const float* __restrict__ h0,
                                                  float* __restrict__ ctx) {
  int idx = blockIdx.x * 256 + threadIdx.x;
  if (idx >= BK_ * D_) return;
  int row = idx / D_;
  int dd  = idx % D_;
  ctx[idx] = tanhf(h0[(row / K_) * D_ + dd]);
}

__global__ __launch_bounds__(256) void init_kernel(int* genA, float* scores,
                                                   float* finals, int* done) {
  int idx = blockIdx.x * 256 + threadIdx.x;
  if (idx < BK_ * GC_) genA[idx] = SOS_T;
  if (idx < BK_ * SC_) scores[idx] = 0.0f;
  if (idx < BK_)       finals[idx] = 0.0f;
  if (idx == 0)        *done = 0;
}

// ---------------- per-step kernels ----------------

// x[row][d] = bf16( emb[tok[row]][d] + ctx[row][d] )
__global__ __launch_bounds__(256) void gather_kernel(const float* __restrict__ emb,
                                                     const float* __restrict__ ctx,
                                                     __bf16* __restrict__ x,
                                                     const int* __restrict__ gen_cur,
                                                     int p) {
  int row = blockIdx.x;
  int tok = gen_cur[row * GC_ + p];
  const float* e = emb + (size_t)tok * D_;
  const float* c = ctx + (size_t)row * D_;
  __bf16* xr = x + (size_t)row * D_;
  for (int dd = threadIdx.x; dd < D_; dd += 256)
    xr[dd] = (__bf16)(e[dd] + c[dd]);
}

// 10 waves per block (one per 16-row M tile), block covers a 64-column N chunk.
// B slab (64 cols x 32 K, bf16 = 4KB) staged through LDS once per K-step,
// consumed by all 10 waves. Async global->LDS when the toolchain exposes it.
__global__ __launch_bounds__(320) void gemm_kernel(const __bf16* __restrict__ x,
                                                   const __bf16* __restrict__ wt,
                                                   float* __restrict__ logits) {
  __shared__ __attribute__((aligned(16))) __bf16 bsh[2][64 * 32]; // [buf][n*32 + k]

  const int tid   = threadIdx.x;
  const int lane  = tid & 31;
  const int wave  = tid >> 5;          // 0..9 == mtile
  const int l15   = lane & 15;
  const int khalf = lane >> 4;
  const int nbase = blockIdx.x * 64;

  const __bf16* xrow = x + (size_t)(wave * 16 + l15) * D_;

  // cooperative stage indices: threads 0..255 move one 16B segment each
  const int sn  = tid >> 2;            // n within chunk (0..63)
  const int seg = tid & 3;             // 16B segment within 64B row

  v8f acc[4];
#pragma unroll
  for (int t = 0; t < 4; ++t) acc[t] = 0.0f;

  // prologue: stage k0 = 0 into buffer 0
  if (tid < 256) {
    const __bf16* src = wt + (size_t)(nbase + sn) * D_ + seg * 8;
#ifdef HAVE_ASYNC_LDS
    __builtin_amdgcn_global_load_async_to_lds_b128(
        ASYNC_PTR(src), ASYNC_PTR(&bsh[0][sn * 32 + seg * 8]), 0, 0);
#else
    *(v8bf*)(&bsh[0][sn * 32 + seg * 8]) = *(const v8bf*)src;
#endif
  }
#ifdef HAVE_ASYNC_LDS
  WAIT_ASYNC();
#endif
  __syncthreads();

  for (int ks = 0; ks < D_ / 32; ++ks) {
    const int k0  = ks * 32;
    const int buf = ks & 1;

    // stage next slab into the other buffer (overlaps with WMMA below)
    if (ks + 1 < D_ / 32 && tid < 256) {
      const __bf16* src = wt + (size_t)(nbase + sn) * D_ + (k0 + 32) + seg * 8;
#ifdef HAVE_ASYNC_LDS
      __builtin_amdgcn_global_load_async_to_lds_b128(
          ASYNC_PTR(src), ASYNC_PTR(&bsh[buf ^ 1][sn * 32 + seg * 8]), 0, 0);
#else
      *(v8bf*)(&bsh[buf ^ 1][sn * 32 + seg * 8]) = *(const v8bf*)src;
#endif
    }

    // A tile from global (tiny, cache-hot)
    v8bf alo = *(const v8bf*)(xrow + k0 + khalf * 16);
    v8bf ahi = *(const v8bf*)(xrow + k0 + khalf * 16 + 8);
    v16bf a = __builtin_shufflevector(alo, ahi, 0,1,2,3,4,5,6,7,8,9,10,11,12,13,14,15);

#pragma unroll
    for (int t = 0; t < 4; ++t) {
      const __bf16* bp = &bsh[buf][(t * 16 + l15) * 32 + khalf * 16];
      v8bf blo = *(const v8bf*)(bp);
      v8bf bhi = *(const v8bf*)(bp + 8);
      v16bf bm = __builtin_shufflevector(blo, bhi, 0,1,2,3,4,5,6,7,8,9,10,11,12,13,14,15);
      acc[t] = __builtin_amdgcn_wmma_f32_16x16x32_bf16(
          false, a, false, bm, (short)0, acc[t], false, false);
    }
#ifdef HAVE_ASYNC_LDS
    WAIT_ASYNC();                      // staged slab resident before publish
#endif
    __syncthreads();
  }

  // C layout: lane -> N = l15, rows m0..m0+7 where m0 = 8*khalf
  const int m0 = wave * 16 + khalf * 8;
#pragma unroll
  for (int t = 0; t < 4; ++t) {
    const int n = nbase + t * 16 + l15;
#pragma unroll
    for (int r = 0; r < 8; ++r)
      logits[(size_t)(m0 + r) * VPAD + n] = acc[t][r];
  }
}

// single-pass online softmax: per-row max and log(sum(exp(x - max)))
__global__ __launch_bounds__(256) void rowreduce_kernel(const float* __restrict__ logits,
                                                        float* __restrict__ rowmax,
                                                        float* __restrict__ rowlz) {
  int row = blockIdx.x;
  int tid = threadIdx.x;
  const float* lr = logits + (size_t)row * VPAD;
  __shared__ float smm[256];
  __shared__ float sms[256];

  float m = NEGINF, s = 0.0f;
  for (int v = tid; v < VSZ; v += 256) {
    float xv = lr[v];
    if (xv > m) { s = s * expf(m - xv) + 1.0f; m = xv; }  // expf(-inf)=0 first iter
    else        { s += expf(xv - m); }
  }
  smm[tid] = m; sms[tid] = s;
  __syncthreads();
  for (int st = 128; st >= 1; st >>= 1) {
    if (tid < st) {
      float m1 = smm[tid], s1 = sms[tid];
      float m2 = smm[tid + st], s2 = sms[tid + st];
      float M = fmaxf(m1, m2);
      smm[tid] = M;
      sms[tid] = s1 * expf(m1 - M) + s2 * expf(m2 - M);
    }
    __syncthreads();
  }
  if (tid == 0) {
    rowmax[row] = smm[0];
    rowlz[row]  = logf(sms[0]);
  }
}

__device__ __forceinline__ void insert5(float* s, int* bm, int* wd,
                                        float val, int beam, int word) {
  if (val <= s[4]) return;
  int t = 4;
  while (t > 0 && s[t - 1] < val) {
    s[t] = s[t - 1]; bm[t] = bm[t - 1]; wd[t] = wd[t - 1];
    --t;
  }
  s[t] = val; bm[t] = beam; wd[t] = word;
}

__device__ __forceinline__ void merge5(float* as, int* ab, int* aw,
                                       const float* bs, const int* bb, const int* bw) {
  float os[5]; int ob[5], ow[5];
  int i = 0, j = 0;
#pragma unroll
  for (int t = 0; t < 5; ++t) {
    bool takeA = (j >= 5) || (i < 5 && as[i] >= bs[j]);
    if (takeA) { os[t] = as[i]; ob[t] = ab[i]; ow[t] = aw[i]; ++i; }
    else       { os[t] = bs[j]; ob[t] = bb[j]; ow[t] = bw[j]; ++j; }
  }
#pragma unroll
  for (int t = 0; t < 5; ++t) { as[t] = os[t]; ab[t] = ob[t]; aw[t] = ow[t]; }
}

// per batch (blockIdx.x = b) top-5 over (k, v)
__global__ __launch_bounds__(256) void topk_kernel(const float* __restrict__ logits,
                                                   const float* __restrict__ rowmax,
                                                   const float* __restrict__ rowlz,
                                                   const int* __restrict__ gen_cur,
                                                   const float* __restrict__ scores,
                                                   float* __restrict__ tops,
                                                   int* __restrict__ topb,
                                                   int* __restrict__ topw,
                                                   int p) {
  int b = blockIdx.x;
  int tid = threadIdx.x;

  float ls[5]; int lb[5], lw[5];
#pragma unroll
  for (int t = 0; t < 5; ++t) { ls[t] = NEGINF; lb[t] = 0; lw[t] = 0; }

  for (int k = 0; k < K_; ++k) {
    if (p == 0 && k > 0) break;               // beams >0 masked at first step
    int bk = b * K_ + k;
    int tok = gen_cur[bk * GC_ + p];
    float prev = (p == 0) ? 0.0f : scores[bk * SC_ + (p - 1)];
    if (tok == EOS_T) {                       // eos_row: only v==EOS, logp=0
      if (tid == 0) insert5(ls, lb, lw, prev, k, EOS_T);
      continue;
    }
    const float* lr = logits + (size_t)bk * VPAD;
    float M = rowmax[bk], LZ = rowlz[bk];
    for (int v = tid; v < VSZ; v += 256) {
      if (v == PAD_T || v == SOS_T) continue; // masked to -inf
      float val = lr[v] - M - LZ + prev;
      insert5(ls, lb, lw, val, k, v);
    }
  }

  __shared__ float ss[256 * 5];
  __shared__ int   sb[256 * 5];
  __shared__ int   sw[256 * 5];
#pragma unroll
  for (int t = 0; t < 5; ++t) {
    ss[tid * 5 + t] = ls[t]; sb[tid * 5 + t] = lb[t]; sw[tid * 5 + t] = lw[t];
  }
  __syncthreads();

  for (int stride = 128; stride >= 1; stride >>= 1) {
    if (tid < stride) {
      float as[5], bs2[5]; int ab[5], aw[5], bb2[5], bw2[5];
#pragma unroll
      for (int t = 0; t < 5; ++t) {
        as[t]  = ss[tid * 5 + t];            ab[t]  = sb[tid * 5 + t];
        aw[t]  = sw[tid * 5 + t];
        bs2[t] = ss[(tid + stride) * 5 + t]; bb2[t] = sb[(tid + stride) * 5 + t];
        bw2[t] = sw[(tid + stride) * 5 + t];
      }
      merge5(as, ab, aw, bs2, bb2, bw2);
#pragma unroll
      for (int t = 0; t < 5; ++t) {
        ss[tid * 5 + t] = as[t]; sb[tid * 5 + t] = ab[t]; sw[tid * 5 + t] = aw[t];
      }
    }
    __syncthreads();
  }

  if (tid < K_) {
    tops[b * K_ + tid] = ss[tid];
    topb[b * K_ + tid] = sb[tid];
    topw[b * K_ + tid] = sw[tid];
  }
}

// beam reindex + score/final update + done flag
__global__ __launch_bounds__(256) void update_kernel(const int* __restrict__ gen_cur,
                                                     int* __restrict__ gen_next,
                                                     float* __restrict__ scores,
                                                     float* __restrict__ finals,
                                                     const float* __restrict__ tops,
                                                     const int* __restrict__ topb,
                                                     const int* __restrict__ topw,
                                                     int* done, int p) {
  __shared__ int eoscnt;
  __shared__ int dsh;
  int tid = threadIdx.x;
  if (tid == 0) { eoscnt = 0; dsh = *done; }
  __syncthreads();
  int d = dsh;

  if (tid < BK_) {
    int b = tid / K_, k = tid % K_;
    int word = topw[b * K_ + k];
    int beam = topb[b * K_ + k];
    float s  = tops[b * K_ + k];
    if (!d) {
      const int* src = gen_cur + (size_t)(b * K_ + beam) * GC_;
      int* dst = gen_next + (size_t)tid * GC_;
      for (int c = 0; c < GC_; ++c) dst[c] = src[c];
      dst[p + 1] = word;
      scores[tid * SC_ + (p + 1)] = s;
      finals[tid] = s;
    } else {
      const int* src = gen_cur + (size_t)tid * GC_;
      int* dst = gen_next + (size_t)tid * GC_;
      for (int c = 0; c < GC_; ++c) dst[c] = src[c];
    }
    if (word == EOS_T) atomicAdd(&eoscnt, 1);
  }
  __syncthreads();
  if (tid == 0 && eoscnt == BK_) *done = 1;
}

__global__ __launch_bounds__(256) void finalize_kernel(int* __restrict__ gen,
                                                       const float* __restrict__ finals,
                                                       const int* __restrict__ done,
                                                       int* __restrict__ out) {
  __shared__ int dsh;
  int tid = threadIdx.x;
  if (tid == 0) dsh = *done;
  __syncthreads();
  if (!dsh) {
    for (int r = tid; r < BK_; r += 256) gen[r * GC_ + L_] = EOS_T;
  }
  __syncthreads();
  if (tid < B_) {
    int best = 0;
    float bs = finals[tid * K_];
    for (int k = 1; k < K_; ++k) {
      float s = finals[tid * K_ + k];
      if (s > bs) { bs = s; best = k; }
    }
    const int* row = gen + (size_t)(tid * K_ + best) * GC_;
    for (int c = 0; c < SC_; ++c) out[tid * SC_ + c] = row[1 + c];   // best_sent[:,1:]
  }
}

// ---------------- host side ----------------
extern "C" void kernel_launch(void* const* d_in, const int* in_sizes, int n_in,
                              void* d_out, int out_size, void* d_ws, size_t ws_size,
                              hipStream_t stream) {
  (void)in_sizes; (void)n_in; (void)out_size; (void)ws_size;
  const float* emb   = (const float*)d_in[0];   // [V][D]
  const float* w_out = (const float*)d_in[1];   // [D][V]
  const float* h0    = (const float*)d_in[2];   // [B][D]

  char* w = (char*)d_ws;
  size_t off = 0;
  auto carve = [&](size_t bytes) -> void* {
    void* p = w + off;
    off = (off + bytes + 255) & ~(size_t)255;
    return p;
  };
  __bf16* wt     = (__bf16*)carve((size_t)VPAD * D_ * sizeof(__bf16));
  float*  logits = (float*) carve((size_t)BK_ * VPAD * sizeof(float));
  float*  ctx    = (float*) carve((size_t)BK_ * D_ * sizeof(float));
  __bf16* xbf    = (__bf16*)carve((size_t)BK_ * D_ * sizeof(__bf16));
  int*    genA   = (int*)   carve((size_t)BK_ * GC_ * sizeof(int));
  int*    genB   = (int*)   carve((size_t)BK_ * GC_ * sizeof(int));
  float*  scores = (float*) carve((size_t)BK_ * SC_ * sizeof(float));
  float*  finals = (float*) carve((size_t)BK_ * sizeof(float));
  float*  rowmax = (float*) carve((size_t)BK_ * sizeof(float));
  float*  rowlz  = (float*) carve((size_t)BK_ * sizeof(float));
  float*  tops   = (float*) carve((size_t)B_ * K_ * sizeof(float));
  int*    topb   = (int*)   carve((size_t)B_ * K_ * sizeof(int));
  int*    topw   = (int*)   carve((size_t)B_ * K_ * sizeof(int));
  int*    done   = (int*)   carve(sizeof(int));

  {
    size_t tot = (size_t)D_ * VPAD;
    int blocks = (int)((tot + 255) / 256);
    convert_wt<<<blocks, 256, 0, stream>>>(w_out, wt);
  }
  ctx_kernel<<<(BK_ * D_ + 255) / 256, 256, 0, stream>>>(h0, ctx);
  init_kernel<<<(BK_ * GC_ + 255) / 256, 256, 0, stream>>>(genA, scores, finals, done);

  for (int p = 0; p < L_ - 1; ++p) {
    const int* gen_cur = (p & 1) ? genB : genA;
    int*       gen_nxt = (p & 1) ? genA : genB;
    gather_kernel<<<BK_, 256, 0, stream>>>(emb, ctx, xbf, gen_cur, p);
    gemm_kernel<<<dim3(VPAD / 64), 320, 0, stream>>>(xbf, wt, logits);
    rowreduce_kernel<<<BK_, 256, 0, stream>>>(logits, rowmax, rowlz);
    topk_kernel<<<B_, 256, 0, stream>>>(logits, rowmax, rowlz, gen_cur, scores,
                                        tops, topb, topw, p);
    update_kernel<<<1, 256, 0, stream>>>(gen_cur, gen_nxt, scores, finals,
                                         tops, topb, topw, done, p);
  }
  // last step p=30 (even) wrote genB
  finalize_kernel<<<1, 256, 0, stream>>>(genB, finals, done, (int*)d_out);
}
